// AttentionBlock_5978594476753
// MI455X (gfx1250) — matrix operand order repaired and checked by reading, not compiled, not run on hardware
//
#include <hip/hip_runtime.h>
#include <hip/hip_bf16.h>
#include <stdint.h>

#define DEV __device__ __forceinline__

typedef __attribute__((ext_vector_type(16))) __bf16 v16bf;
typedef __attribute__((ext_vector_type(8)))  float  v8f;

union Frag {
    uint4 q[2];
    v16bf v;
};

#define WMMA_BF16(A, B, C)                                                     \
    __builtin_amdgcn_wmma_f32_16x16x32_bf16(false, (A).v, false, (B).v,        \
                                            (short)0, (C), false, false)

DEV uint4 ldg4(const uint16_t* p) { return *reinterpret_cast<const uint4*>(p); }
DEV uint4 lds4(const uint16_t* p) { return *reinterpret_cast<const uint4*>(p); }

DEV uint16_t f2bf(float f) {
    uint32_t u = __float_as_uint(f);
    u += 0x7FFFu + ((u >> 16) & 1u);   // round-to-nearest-even
    return (uint16_t)(u >> 16);
}

// CDNA5 async copy: 16B per lane, global -> LDS, tracked by ASYNCcnt.
// lds_off = wave-relative LDS byte offset (generic shared ptr truncated to 32b,
// per ISA flat->LDS mapping addr[31:0]).
DEV void async_g2l_b128(uint32_t lds_off, const void* gaddr) {
    asm volatile("global_load_async_to_lds_b128 %0, %1, off"
                 :: "v"(lds_off), "v"(gaddr)
                 : "memory");
}
DEV void wait_async0() { asm volatile("s_wait_asynccnt 0x0" ::: "memory"); }

// cross-lane reductions within each 16-lane half of the wave32 (xor 1,2,4,8
// never flips bit4, so the two halves stay independent)
DEV float red_max16(float v) {
    int l = (int)(threadIdx.x & 31u);
#pragma unroll
    for (int m = 1; m < 16; m <<= 1) {
        float o = __int_as_float(
            __builtin_amdgcn_ds_bpermute(((l ^ m) << 2), __float_as_int(v)));
        v = fmaxf(v, o);
    }
    return v;
}
DEV float red_add16(float v) {
    int l = (int)(threadIdx.x & 31u);
#pragma unroll
    for (int m = 1; m < 16; m <<= 1) {
        v += __int_as_float(
            __builtin_amdgcn_ds_bpermute(((l ^ m) << 2), __float_as_int(v)));
    }
    return v;
}

// ---------------------------------------------------------------- convert ---
__global__ __launch_bounds__(256) void k_cvt(const float* __restrict__ s,
                                             uint16_t* __restrict__ d, int n) {
    int i = (blockIdx.x * 256 + threadIdx.x) * 4;
    if (i + 3 < n) {
        float4 f = *reinterpret_cast<const float4*>(s + i);
        d[i + 0] = f2bf(f.x);
        d[i + 1] = f2bf(f.y);
        d[i + 2] = f2bf(f.z);
        d[i + 3] = f2bf(f.w);
    }
}

// ------------------------------------------------------- QKV projections ----
// C[8192,512] = xb[8192,512] @ W^T (+bias). 64x64 tile per workgroup,
// 8 waves * two 16x16 WMMA tiles. mode: 0=Q (scaled 1/8, [B,H,S,DH]),
// 1=K ([B,H,S,DH]), 2=V transposed ([B,H,DH,S]).
__global__ __launch_bounds__(256) void k_qkv(const uint16_t* __restrict__ xb,
                                             const uint16_t* __restrict__ wb,
                                             const float* __restrict__ bias,
                                             uint16_t* __restrict__ out,
                                             int mode) {
    const int w = threadIdx.x >> 5, lane = threadIdx.x & 31;
    const int half = lane >> 4, l16 = lane & 15;
    const int rowBase = blockIdx.x * 64 + ((w & 3) << 4);
    const int colBase = blockIdx.y * 64 + ((w >> 2) << 5);

    v8f acc0 = {}, acc1 = {};
    const uint16_t* arow = xb + (size_t)(rowBase + l16) * 512;
    const uint16_t* w0 = wb + (size_t)(colBase + l16) * 512;
    const uint16_t* w1 = wb + (size_t)(colBase + 16 + l16) * 512;

    for (int k0 = 0; k0 < 512; k0 += 32) {
        Frag a, b0, b1;
        const int ak = k0 + half * 8;          // A: lanes<16 K{0..7,16..23}
        a.q[0] = ldg4(arow + ak);
        a.q[1] = ldg4(arow + ak + 16);
        const int bk = k0 + half * 8;          // B: lanes<16 K{0..7,16..23}
        b0.q[0] = ldg4(w0 + bk);
        b0.q[1] = ldg4(w0 + bk + 16);
        b1.q[0] = ldg4(w1 + bk);
        b1.q[1] = ldg4(w1 + bk + 16);
        if (k0 + 32 < 512) {
            __builtin_prefetch(arow + ak + 32, 0, 3);
            __builtin_prefetch(w0 + bk + 32, 0, 3);
            __builtin_prefetch(w1 + bk + 32, 0, 3);
        }
        acc0 = WMMA_BF16(a, b0, acc0);
        acc1 = WMMA_BF16(a, b1, acc1);
    }

    const float scale = (mode == 0) ? 0.125f : 1.0f;  // 1/sqrt(64) folded in Q
#pragma unroll
    for (int t = 0; t < 2; ++t) {
        v8f acc = t ? acc1 : acc0;
        const int j = colBase + t * 16 + l16;
        const float bv = bias[j];
        const int h = j >> 6, dh = j & 63;
#pragma unroll
        for (int r = 0; r < 8; ++r) {
            const int tok = rowBase + r + half * 8;
            const int b = tok >> 11, s = tok & 2047;
            const uint16_t bf = f2bf((acc[r] + bv) * scale);
            if (mode == 2)
                out[(size_t)((b * 8 + h) * 64 + dh) * 2048 + s] = bf;   // V^T
            else
                out[(size_t)((b * 8 + h) * 2048 + s) * 64 + dh] = bf;   // Q,K
        }
    }
}

// ----------------------------------------------------- flash attention ------
// grid: (qblock 0..15, bh 0..31). 8 waves * 16 queries = 128 queries/block.
// K/V tiles staged cooperatively into LDS via async-to-LDS DMA, double
// buffered: one barrier per 32-key block, DMA for block i+1 overlaps compute.
__global__ __launch_bounds__(256) void k_attn(const uint16_t* __restrict__ qb,
                                              const uint16_t* __restrict__ kb,
                                              const uint16_t* __restrict__ vt,
                                              uint16_t* __restrict__ ao) {
    __shared__ __align__(16) uint16_t pls[8 * 16 * 32];   // per-wave P slabs
    __shared__ __align__(16) uint16_t ldsK[2][32 * 64];   // [key][dh]
    __shared__ __align__(16) uint16_t ldsV[2][64 * 32];   // [dh][key]
    const int w = threadIdx.x >> 5, lane = threadIdx.x & 31;
    const int half = lane >> 4, l16 = lane & 15;
    const int bh = blockIdx.y;
    const int q0 = blockIdx.x * 128 + w * 16;

    // Q fragments (stay in registers for whole kernel); scale pre-folded.
    Frag qa0, qa1;
    {
        const uint16_t* qrow = qb + ((size_t)bh * 2048 + q0 + l16) * 64;
        const int k1 = half * 8;
        qa0.q[0] = ldg4(qrow + k1);
        qa0.q[1] = ldg4(qrow + k1 + 16);
        qa1.q[0] = ldg4(qrow + 32 + k1);
        qa1.q[1] = ldg4(qrow + 32 + k1 + 16);
    }

    v8f o0 = {}, o1 = {}, o2 = {}, o3 = {};
    float m_[8], l_[8];
#pragma unroll
    for (int r = 0; r < 8; ++r) { m_[r] = -3.0e38f; l_[r] = 0.f; }

    const uint16_t* kbase = kb + (size_t)bh * 2048 * 64;
    const uint16_t* vbase = vt + (size_t)bh * 64 * 2048;
    uint16_t* pw = pls + w * 512;

    // Per-thread staging slices: 256 threads x 16B == one 4KB tile each.
    const int tid = threadIdx.x;
    const int kkey = tid >> 3, kchunk = tid & 7;   // K: 32 rows x 8 chunks
    const int vdh = tid >> 2, vchunk = tid & 3;    // V: 64 rows x 4 chunks
    const uint32_t koff[2] = {
        (uint32_t)(uintptr_t)&ldsK[0][kkey * 64 + kchunk * 8],
        (uint32_t)(uintptr_t)&ldsK[1][kkey * 64 + kchunk * 8]};
    const uint32_t voff[2] = {
        (uint32_t)(uintptr_t)&ldsV[0][vdh * 32 + vchunk * 8],
        (uint32_t)(uintptr_t)&ldsV[1][vdh * 32 + vchunk * 8]};
    const uint16_t* kgsrc = kbase + (size_t)kkey * 64 + kchunk * 8;
    const uint16_t* vgsrc = vbase + (size_t)vdh * 2048 + vchunk * 8;

    // prologue: stage block 0 into buffer 0
    async_g2l_b128(koff[0], kgsrc);
    async_g2l_b128(voff[0], vgsrc);

    for (int it = 0; it < 64; ++it) {
        const int buf = it & 1;
        wait_async0();        // this thread's DMA for `buf` is in LDS
        __syncthreads();      // everyone's DMA landed; prior reads of it done

        // stage next block into the other buffer while we compute this one
        if (it + 1 < 64) {
            const int nk = (it + 1) * 32;
            async_g2l_b128(koff[buf ^ 1], kgsrc + (size_t)nk * 64);
            async_g2l_b128(voff[buf ^ 1], vgsrc + nk);
        }

        // K^T fragments from LDS: B[k=dh, n=key]; lane = key, table-order dh.
        Frag bk00, bk01, bk10, bk11;
        {
            const uint16_t* kr0 = &ldsK[buf][l16 * 64 + half * 8];
            const uint16_t* kr1 = kr0 + 16 * 64;
            bk00.q[0] = lds4(kr0);      bk00.q[1] = lds4(kr0 + 16);
            bk01.q[0] = lds4(kr0 + 32); bk01.q[1] = lds4(kr0 + 48);
            bk10.q[0] = lds4(kr1);      bk10.q[1] = lds4(kr1 + 16);
            bk11.q[0] = lds4(kr1 + 32); bk11.q[1] = lds4(kr1 + 48);
        }
        v8f s0 = {}, s1 = {};
        s0 = WMMA_BF16(qa0, bk00, s0);
        s0 = WMMA_BF16(qa1, bk01, s0);
        s1 = WMMA_BF16(qa0, bk10, s1);
        s1 = WMMA_BF16(qa1, bk11, s1);

        // streaming softmax update (per C-tile row r; rows = r + 8*half)
        float c_[8];
#pragma unroll
        for (int r = 0; r < 8; ++r) {
            float mx = red_max16(fmaxf(s0[r], s1[r]));
            float mn = fmaxf(m_[r], mx);
            c_[r] = __expf(m_[r] - mn);
            m_[r] = mn;
        }
#pragma unroll
        for (int r = 0; r < 8; ++r) {
            float p0 = __expf(s0[r] - m_[r]);
            float p1 = __expf(s1[r] - m_[r]);
            float rs = red_add16(p0 + p1);
            l_[r] = l_[r] * c_[r] + rs;
            o0[r] *= c_[r]; o1[r] *= c_[r]; o2[r] *= c_[r]; o3[r] *= c_[r];
            const int row = r + half * 8;
            pw[row * 32 + l16]      = f2bf(p0);   // transpose P via LDS
            pw[row * 32 + 16 + l16] = f2bf(p1);
        }
        asm volatile("s_wait_dscnt 0x0" ::: "memory");  // wave-ordered DS RAW

        Frag pa;  // P as A-fragment (16 queries x 32 keys)
        {
            const uint16_t* pr = pw + l16 * 32 + half * 8;
            pa.q[0] = lds4(pr);
            pa.q[1] = lds4(pr + 16);
        }
        // V fragments from LDS V^T: B[k=key, n=dh]; lane = dh, table-order keys.
        const uint16_t* vr = &ldsV[buf][l16 * 32 + half * 8];
        Frag v0f, v1f, v2f, v3f;
        v0f.q[0] = lds4(vr);           v0f.q[1] = lds4(vr + 16);
        v1f.q[0] = lds4(vr + 16 * 32); v1f.q[1] = lds4(vr + 16 * 32 + 16);
        v2f.q[0] = lds4(vr + 32 * 32); v2f.q[1] = lds4(vr + 32 * 32 + 16);
        v3f.q[0] = lds4(vr + 48 * 32); v3f.q[1] = lds4(vr + 48 * 32 + 16);
        o0 = WMMA_BF16(pa, v0f, o0);
        o1 = WMMA_BF16(pa, v1f, o1);
        o2 = WMMA_BF16(pa, v2f, o2);
        o3 = WMMA_BF16(pa, v3f, o3);
    }

    const int b = bh >> 3, h = bh & 7;
#pragma unroll
    for (int r = 0; r < 8; ++r) {
        const int s = q0 + r + half * 8;
        const float inv = 1.0f / l_[r];
        const size_t base = ((size_t)(b * 2048 + s)) * 512 + h * 64;
        ao[base + l16 +  0] = f2bf(o0[r] * inv);
        ao[base + l16 + 16] = f2bf(o1[r] * inv);
        ao[base + l16 + 32] = f2bf(o2[r] * inv);
        ao[base + l16 + 48] = f2bf(o3[r] * inv);
    }
}

// --------------------------------- O-projection + residual + LayerNorm ------
// grid: 512 blocks; block = 16 tokens x full 512 cols (wave w -> cols w*64..).
__global__ __launch_bounds__(256) void k_oln(const uint16_t* __restrict__ ab,
                                             const uint16_t* __restrict__ wob,
                                             const float* __restrict__ bo,
                                             const float* __restrict__ x,
                                             const float* __restrict__ gamma,
                                             const float* __restrict__ beta,
                                             float* __restrict__ out) {
    __shared__ float red[2][16][8];
    __shared__ float stat[2][16];
    const int w = threadIdx.x >> 5, lane = threadIdx.x & 31;
    const int half = lane >> 4, l16 = lane & 15;
    const int t0 = blockIdx.x * 16;
    const int n0 = w * 64;

    v8f a0 = {}, a1 = {}, a2 = {}, a3 = {};
    const uint16_t* arow = ab + (size_t)(t0 + l16) * 512;
    const uint16_t* wj0 = wob + (size_t)(n0 + l16) * 512;
    const uint16_t* wj1 = wj0 + 16 * 512;
    const uint16_t* wj2 = wj0 + 32 * 512;
    const uint16_t* wj3 = wj0 + 48 * 512;
    for (int k0 = 0; k0 < 512; k0 += 32) {
        Frag a, b0, b1, b2, b3;
        const int ak = k0 + half * 8;
        a.q[0] = ldg4(arow + ak);
        a.q[1] = ldg4(arow + ak + 16);
        const int bk = k0 + half * 8;
        b0.q[0] = ldg4(wj0 + bk); b0.q[1] = ldg4(wj0 + bk + 16);
        b1.q[0] = ldg4(wj1 + bk); b1.q[1] = ldg4(wj1 + bk + 16);
        b2.q[0] = ldg4(wj2 + bk); b2.q[1] = ldg4(wj2 + bk + 16);
        b3.q[0] = ldg4(wj3 + bk); b3.q[1] = ldg4(wj3 + bk + 16);
        if (k0 + 32 < 512) __builtin_prefetch(arow + ak + 32, 0, 3);
        a0 = WMMA_BF16(a, b0, a0);
        a1 = WMMA_BF16(a, b1, a1);
        a2 = WMMA_BF16(a, b2, a2);
        a3 = WMMA_BF16(a, b3, a3);
    }

    float y[4][8], ps[8], psq[8];
#pragma unroll
    for (int r = 0; r < 8; ++r) { ps[r] = 0.f; psq[r] = 0.f; }
#pragma unroll
    for (int t = 0; t < 4; ++t) {
        const int j = n0 + t * 16 + l16;
        const float bv = bo[j];
        v8f acc = (t == 0) ? a0 : (t == 1) ? a1 : (t == 2) ? a2 : a3;
#pragma unroll
        for (int r = 0; r < 8; ++r) {
            const int tok = t0 + r + half * 8;
            const float v = acc[r] + bv + x[(size_t)tok * 512 + j];
            y[t][r] = v;
            ps[r] += v;
            psq[r] += v * v;
        }
    }
#pragma unroll
    for (int r = 0; r < 8; ++r) {
        ps[r] = red_add16(ps[r]);
        psq[r] = red_add16(psq[r]);
    }
    if (l16 == 0) {
#pragma unroll
        for (int r = 0; r < 8; ++r) {
            red[0][r + half * 8][w] = ps[r];
            red[1][r + half * 8][w] = psq[r];
        }
    }
    __syncthreads();
    if (threadIdx.x < 16) {
        float s = 0.f, sq = 0.f;
#pragma unroll
        for (int i = 0; i < 8; ++i) {
            s += red[0][threadIdx.x][i];
            sq += red[1][threadIdx.x][i];
        }
        const float mu = s * (1.0f / 512.0f);
        const float var = sq * (1.0f / 512.0f) - mu * mu;
        stat[0][threadIdx.x] = mu;
        stat[1][threadIdx.x] = rsqrtf(var + 1e-5f);
    }
    __syncthreads();
#pragma unroll
    for (int t = 0; t < 4; ++t) {
        const int j = n0 + t * 16 + l16;
        const float g = gamma[j], bt = beta[j];
#pragma unroll
        for (int r = 0; r < 8; ++r) {
            const int row = r + half * 8;
            out[(size_t)(t0 + row) * 512 + j] =
                (y[t][r] - stat[0][row]) * stat[1][row] * g + bt;
        }
    }
}

// ----------------------------------------------------------------- launch ---
extern "C" void kernel_launch(void* const* d_in, const int* in_sizes, int n_in,
                              void* d_out, int out_size, void* d_ws,
                              size_t ws_size, hipStream_t stream) {
    const float* x  = (const float*)d_in[0];
    const float* Wq = (const float*)d_in[1];
    const float* bq = (const float*)d_in[2];
    const float* Wk = (const float*)d_in[3];
    const float* bk = (const float*)d_in[4];
    const float* Wv = (const float*)d_in[5];
    const float* bv = (const float*)d_in[6];
    const float* Wo = (const float*)d_in[7];
    const float* bo = (const float*)d_in[8];
    const float* gamma = (const float*)d_in[9];
    const float* beta  = (const float*)d_in[10];
    float* out = (float*)d_out;

    uint8_t* ws = (uint8_t*)d_ws;
    size_t off = 0;
    auto alloc = [&](size_t elems) {
        uint16_t* p = (uint16_t*)(ws + off);
        off += elems * sizeof(uint16_t);
        return p;
    };
    const size_t NTOK = 8192, D = 512;
    uint16_t* xb  = alloc(NTOK * D);
    uint16_t* wqb = alloc(D * D);
    uint16_t* wkb = alloc(D * D);
    uint16_t* wvb = alloc(D * D);
    uint16_t* wob = alloc(D * D);
    uint16_t* qbuf = alloc(NTOK * D);
    uint16_t* kbuf = alloc(NTOK * D);
    uint16_t* vtb  = alloc(NTOK * D);
    uint16_t* abuf = alloc(NTOK * D);
    (void)ws_size; (void)n_in; (void)in_sizes; (void)out_size;

    k_cvt<<<(int)(NTOK * D / 1024), 256, 0, stream>>>(x, xb, (int)(NTOK * D));
    k_cvt<<<(int)(D * D / 1024), 256, 0, stream>>>(Wq, wqb, (int)(D * D));
    k_cvt<<<(int)(D * D / 1024), 256, 0, stream>>>(Wk, wkb, (int)(D * D));
    k_cvt<<<(int)(D * D / 1024), 256, 0, stream>>>(Wv, wvb, (int)(D * D));
    k_cvt<<<(int)(D * D / 1024), 256, 0, stream>>>(Wo, wob, (int)(D * D));

    dim3 gq(128, 8);
    k_qkv<<<gq, 256, 0, stream>>>(xb, wqb, bq, qbuf, 0);
    k_qkv<<<gq, 256, 0, stream>>>(xb, wkb, bk, kbuf, 1);
    k_qkv<<<gq, 256, 0, stream>>>(xb, wvb, bv, vtb, 2);

    k_attn<<<dim3(16, 32), 256, 0, stream>>>(qbuf, kbuf, vtb, abuf);

    k_oln<<<512, 256, 0, stream>>>(abuf, wob, bo, x, gamma, beta, out);
}
